// Window_Attention_46583215292996
// MI455X (gfx1250) — matrix operand order repaired and checked by reading, not compiled, not run on hardware
//
#include <hip/hip_runtime.h>
#include <hip/hip_bf16.h>
#include <math.h>

// ---------------------------------------------------------------------------
// Problem constants
// ---------------------------------------------------------------------------
#define BATCH   32
#define HH      64
#define WW      64
#define CC      192
#define WS      8
#define NH      6
#define HD      32          // head dim
#define HID     768
#define NTOK    (BATCH * HH * WW)     // 131072 tokens
#define WTOK    (WS*WS)               // 64 tokens per window
#define QKVN    (3*CC)                // 576
#define ATT_SCALE 0.17677669529663687f  // 32^-0.5

typedef __attribute__((ext_vector_type(16))) __bf16        v16bf;
typedef __attribute__((ext_vector_type(8)))  float         v8f;
typedef __attribute__((ext_vector_type(4)))  unsigned int  u32x4;

union FragU {
    v16bf v;
    u32x4 q[2];
};

static __device__ __forceinline__ v8f wmma_bf16(v16bf a, v16bf b, v8f c) {
    // D = A(16x32 bf16) x B(32x16 bf16) + C(16x16 f32)
    return __builtin_amdgcn_wmma_f32_16x16x32_bf16(
        /*neg_a*/false, a, /*neg_b*/false, b,
        /*c_mod*/(short)0, c, /*reuse_a*/false, /*reuse_b*/false);
}

// raster token index -> window-major token index
static __device__ __forceinline__ int raster_to_winmajor(int t) {
    int b   = t >> 12;          // 4096 tokens / image
    int rem = t & 4095;
    int h   = rem >> 6;
    int w   = rem & 63;
    int wi  = ((h >> 3) << 3) + (w >> 3);   // window id 0..63
    int ni  = ((h & 7) << 3) + (w & 7);     // in-window id 0..63
    return (b << 12) + (wi << 6) + ni;
}

// window-major token index -> raster token index
static __device__ __forceinline__ int winmajor_to_raster(int m) {
    int b   = m >> 12;
    int rem = m & 4095;
    int wi  = rem >> 6;
    int ni  = rem & 63;
    int h   = ((wi >> 3) << 3) + (ni >> 3);
    int w   = ((wi & 7) << 3) + (ni & 7);
    return (b << 12) + (h << 6) + w;
}

// ---------------------------------------------------------------------------
// Weight prep: f32 (K x N, row-major) -> bf16 (N x K)  (transpose + convert)
// ---------------------------------------------------------------------------
__global__ void transpose_to_bf16(const float* __restrict__ src,
                                  __bf16* __restrict__ dst, int K, int N) {
    int idx = blockIdx.x * 256 + threadIdx.x;
    if (idx >= K * N) return;
    int n = idx / K;
    int k = idx - n * K;
    dst[(size_t)n * K + k] = (__bf16)src[(size_t)k * N + n];
}

// ---------------------------------------------------------------------------
// LayerNorm over C=192.  One wave (32 lanes) per token, 6 values per lane.
// WINDOW_SCATTER: write output in window-major order (fused window_partition)
// ---------------------------------------------------------------------------
template <bool WINDOW_SCATTER>
__global__ __launch_bounds__(256) void ln_kernel(const float* __restrict__ x,
                                                 const float* __restrict__ g,
                                                 const float* __restrict__ b,
                                                 __bf16* __restrict__ out) {
    int lane = threadIdx.x & 31;
    int wave = threadIdx.x >> 5;
    int t    = blockIdx.x * 8 + wave;           // raster token
    const float* xp = x + (size_t)t * CC;

    float v[6];
    float s = 0.f;
#pragma unroll
    for (int j = 0; j < 6; ++j) { v[j] = xp[lane + 32 * j]; s += v[j]; }
#pragma unroll
    for (int off = 16; off > 0; off >>= 1) s += __shfl_xor(s, off, 32);
    float mean = s * (1.f / 192.f);

    float s2 = 0.f;
#pragma unroll
    for (int j = 0; j < 6; ++j) { float d = v[j] - mean; s2 += d * d; }
#pragma unroll
    for (int off = 16; off > 0; off >>= 1) s2 += __shfl_xor(s2, off, 32);
    float rstd = rsqrtf(s2 * (1.f / 192.f) + 1e-5f);

    int orow = WINDOW_SCATTER ? raster_to_winmajor(t) : t;
    __bf16* op = out + (size_t)orow * CC;
#pragma unroll
    for (int j = 0; j < 6; ++j) {
        int c = lane + 32 * j;
        op[c] = (__bf16)((v[j] - mean) * rstd * g[c] + b[c]);
    }
}

// ---------------------------------------------------------------------------
// bf16 WMMA GEMM:  out(MxN) = A(MxK) * Bt(NxK)^T + bias
//   block = 256 threads = 8 waves arranged 4(M) x 2(N)
//   wave tile 64x32 (4 M-subtiles x 2 N-subtiles), block tile 256x64
//   K is a compile-time constant -> fully unrolled K loop
// Epilogues:
//   EPI_BF16   : store bf16
//   EPI_PROJ   : f32, scatter window-major->raster, += res[raster]
//   EPI_GELU   : exact gelu, store bf16
//   EPI_ADDF32 : f32, += res[m]
// ---------------------------------------------------------------------------
enum { EPI_BF16 = 0, EPI_PROJ = 1, EPI_GELU = 2, EPI_ADDF32 = 3 };

template <int EPI, int K>
__global__ __launch_bounds__(256) void gemm_bf16_wmma(
        const __bf16* __restrict__ A, const __bf16* __restrict__ Bt,
        const float* __restrict__ bias, const float* __restrict__ res,
        void* __restrict__ outv, int N) {
    const int lane = threadIdx.x & 31;
    const int wv   = threadIdx.x >> 5;
    const int wr   = wv >> 1;                 // 0..3
    const int wc   = wv & 1;                  // 0..1
    const int l15  = lane & 15;
    const int lh   = lane >> 4;               // 0 or 1
    const int tm0  = blockIdx.x * 256 + wr * 64;
    const int tn0  = blockIdx.y * 64  + wc * 32;

    const int kbA = lh * 16;   // A frag: contiguous 16 at kbA
    const int kbB = lh * 16;   // B frag: contiguous 16 at kbB

    const __bf16* arow[4];
#pragma unroll
    for (int i = 0; i < 4; ++i)
        arow[i] = A + (size_t)(tm0 + i * 16 + l15) * K + kbA;
    const __bf16* brow[2];
#pragma unroll
    for (int j = 0; j < 2; ++j)
        brow[j] = Bt + (size_t)(tn0 + j * 16 + l15) * K + kbB;

    v8f acc[4][2] = {};

#pragma unroll
    for (int kk = 0; kk < K; kk += 32) {
        FragU a[4], b[2];
#pragma unroll
        for (int i = 0; i < 4; ++i) {
            a[i].q[0] = *(const u32x4*)(arow[i] + kk);
            a[i].q[1] = *(const u32x4*)(arow[i] + kk + 8);
        }
#pragma unroll
        for (int j = 0; j < 2; ++j) {
            b[j].q[0] = *(const u32x4*)(brow[j] + kk);
            b[j].q[1] = *(const u32x4*)(brow[j] + kk + 8);
        }
#pragma unroll
        for (int i = 0; i < 4; ++i)
#pragma unroll
            for (int j = 0; j < 2; ++j)
                acc[i][j] = wmma_bf16(a[i].v, b[j].v, acc[i][j]);
    }

#pragma unroll
    for (int i = 0; i < 4; ++i) {
#pragma unroll
        for (int j = 0; j < 2; ++j) {
            int n     = tn0 + j * 16 + l15;
            float bia = bias[n];
            int mb    = tm0 + i * 16 + lh * 8;
#pragma unroll
            for (int r = 0; r < 8; ++r) {
                int m = mb + r;
                float val = acc[i][j][r] + bia;
                if (EPI == EPI_BF16) {
                    ((__bf16*)outv)[(size_t)m * N + n] = (__bf16)val;
                } else if (EPI == EPI_PROJ) {
                    int t = winmajor_to_raster(m);
                    ((float*)outv)[(size_t)t * N + n] =
                        res[(size_t)t * N + n] + val;
                } else if (EPI == EPI_GELU) {
                    val = 0.5f * val * (1.0f + erff(val * 0.70710678118654752f));
                    ((__bf16*)outv)[(size_t)m * N + n] = (__bf16)val;
                } else { // EPI_ADDF32
                    ((float*)outv)[(size_t)m * N + n] =
                        res[(size_t)m * N + n] + val;
                }
            }
        }
    }
}

// ---------------------------------------------------------------------------
// Window attention: one block (128 thr = 4 waves) per (window, head).
//   S = (q*scale) @ k^T  (64x64, K=32: one WMMA per 16x16 tile)
//   P = softmax(S);  O = P @ v (K=64: two WMMA steps)
// ---------------------------------------------------------------------------
__global__ __launch_bounds__(128) void attn_kernel(
        const __bf16* __restrict__ qkv,   // (NTOK window-major, 576) bf16
        __bf16* __restrict__ attn_out) {  // (NTOK window-major, 192) bf16
    __shared__ __bf16 qs[WTOK * HD];      // 64x32
    __shared__ __bf16 ks[WTOK * HD];      // 64x32
    __shared__ __bf16 vsT[HD * WTOK];     // 32x64 (transposed)
    __shared__ float  S[WTOK * WTOK];     // 64x64
    __shared__ __bf16 P[WTOK * WTOK];     // 64x64

    const int wh   = blockIdx.x;
    const int win  = wh / NH;
    const int head = wh - win * NH;
    const int tid  = threadIdx.x;
    const size_t base = (size_t)win * WTOK;     // first token row of window
    const int coff = head * HD;

    // ---- stage q (scaled), k, v^T into LDS; 16B-vector global loads -------
    // 64 rows x 4 chunks of 8 bf16 = 256 chunks per matrix
    for (int ch = tid; ch < 256; ch += 128) {
        int n  = ch >> 2;          // token 0..63
        int c8 = (ch & 3) << 3;    // chunk channel base
        const __bf16* rowp = qkv + (base + n) * QKVN + coff;

        union { u32x4 q; __bf16 e[8]; } uq, uv;
        uq.q = *(const u32x4*)(rowp + c8);
#pragma unroll
        for (int i = 0; i < 8; ++i)
            uq.e[i] = (__bf16)((float)uq.e[i] * ATT_SCALE);
        *(u32x4*)(qs + n * HD + c8) = uq.q;

        *(u32x4*)(ks + n * HD + c8) = *(const u32x4*)(rowp + CC + c8);

        uv.q = *(const u32x4*)(rowp + 2 * CC + c8);
#pragma unroll
        for (int i = 0; i < 8; ++i)
            vsT[(c8 + i) * WTOK + n] = uv.e[i];   // transpose into LDS
    }
    __syncthreads();

    const int wv   = tid >> 5;     // wave 0..3 -> S row tile
    const int lane = tid & 31;
    const int l15  = lane & 15;
    const int lh   = lane >> 4;
    const int tm   = wv * 16;

    // ---- S = q @ k^T ------------------------------------------------------
    {
        FragU a;
        const __bf16* ap = qs + (tm + l15) * HD + lh * 16;
        a.q[0] = *(const u32x4*)(ap);
        a.q[1] = *(const u32x4*)(ap + 8);
#pragma unroll
        for (int j = 0; j < 4; ++j) {
            FragU b;
            const __bf16* bp = ks + (j * 16 + l15) * HD + lh * 16;
            b.q[0] = *(const u32x4*)(bp);
            b.q[1] = *(const u32x4*)(bp + 8);
            v8f c = {};
            c = wmma_bf16(a.v, b.v, c);
#pragma unroll
            for (int r = 0; r < 8; ++r)
                S[(tm + lh * 8 + r) * WTOK + j * 16 + l15] = c[r];
        }
    }
    __syncthreads();

    // ---- softmax rows (threads 0..63, one row each) -----------------------
    if (tid < WTOK) {
        float* srow = S + tid * WTOK;
        float mx = -3.4e38f;
#pragma unroll 4
        for (int c = 0; c < WTOK; ++c) mx = fmaxf(mx, srow[c]);
        float sum = 0.f;
#pragma unroll 4
        for (int c = 0; c < WTOK; ++c) { float e = expf(srow[c] - mx); srow[c] = e; sum += e; }
        float inv = 1.f / sum;
        __bf16* prow = P + tid * WTOK;
#pragma unroll 4
        for (int c = 0; c < WTOK; ++c) prow[c] = (__bf16)(srow[c] * inv);
    }
    __syncthreads();

    // ---- O = P @ v  (rows tm..tm+15, cols 0..31) --------------------------
#pragma unroll
    for (int j = 0; j < 2; ++j) {
        v8f acc = {};
#pragma unroll
        for (int kk = 0; kk < WTOK; kk += 32) {
            FragU a, b;
            const __bf16* ap = P + (tm + l15) * WTOK + kk + lh * 16;
            a.q[0] = *(const u32x4*)(ap);
            a.q[1] = *(const u32x4*)(ap + 8);
            const __bf16* bp = vsT + (j * 16 + l15) * WTOK + kk + lh * 16;
            b.q[0] = *(const u32x4*)(bp);
            b.q[1] = *(const u32x4*)(bp + 8);
            acc = wmma_bf16(a.v, b.v, acc);
        }
#pragma unroll
        for (int r = 0; r < 8; ++r) {
            attn_out[(base + tm + lh * 8 + r) * CC + coff + j * 16 + l15] =
                (__bf16)acc[r];
        }
    }
}

// ---------------------------------------------------------------------------
// Launch
// ---------------------------------------------------------------------------
extern "C" void kernel_launch(void* const* d_in, const int* in_sizes, int n_in,
                              void* d_out, int out_size, void* d_ws, size_t ws_size,
                              hipStream_t stream) {
    const float* x     = (const float*)d_in[0];
    const float* g1    = (const float*)d_in[1];
    const float* b1    = (const float*)d_in[2];
    const float* wqkv  = (const float*)d_in[3];
    const float* bqkv  = (const float*)d_in[4];
    const float* wproj = (const float*)d_in[5];
    const float* bproj = (const float*)d_in[6];
    const float* g2    = (const float*)d_in[7];
    const float* b2    = (const float*)d_in[8];
    const float* w1    = (const float*)d_in[9];
    const float* bfc1  = (const float*)d_in[10];
    const float* w2    = (const float*)d_in[11];
    const float* bfc2  = (const float*)d_in[12];
    float* out = (float*)d_out;

    // ---- workspace carve --------------------------------------------------
    char* ws = (char*)d_ws;
    size_t off = 0;
    auto carve = [&](size_t bytes) {
        char* p = ws + off;
        off += (bytes + 255) & ~(size_t)255;
        return p;
    };
    __bf16* xln    = (__bf16*)carve((size_t)NTOK * CC   * 2);
    __bf16* qkv    = (__bf16*)carve((size_t)NTOK * QKVN * 2);
    __bf16* attn   = (__bf16*)carve((size_t)NTOK * CC   * 2);
    float*  y      = (float*) carve((size_t)NTOK * CC   * 4);
    __bf16* ln2b   = (__bf16*)carve((size_t)NTOK * CC   * 2);
    __bf16* hid    = (__bf16*)carve((size_t)NTOK * HID  * 2);
    __bf16* wqkvT  = (__bf16*)carve((size_t)CC * QKVN * 2);
    __bf16* wprojT = (__bf16*)carve((size_t)CC * CC   * 2);
    __bf16* w1T    = (__bf16*)carve((size_t)CC * HID  * 2);
    __bf16* w2T    = (__bf16*)carve((size_t)HID * CC  * 2);

    // ---- weights: transpose + bf16 convert --------------------------------
    transpose_to_bf16<<<(CC * QKVN + 255) / 256, 256, 0, stream>>>(wqkv,  wqkvT,  CC,  QKVN);
    transpose_to_bf16<<<(CC * CC   + 255) / 256, 256, 0, stream>>>(wproj, wprojT, CC,  CC);
    transpose_to_bf16<<<(CC * HID  + 255) / 256, 256, 0, stream>>>(w1,    w1T,    CC,  HID);
    transpose_to_bf16<<<(HID * CC  + 255) / 256, 256, 0, stream>>>(w2,    w2T,    HID, CC);

    // ---- LN1 + window partition ------------------------------------------
    ln_kernel<true><<<NTOK / 8, 256, 0, stream>>>(x, g1, b1, xln);

    // ---- QKV GEMM: (M x 192) @ (192 x 576) -> bf16 ------------------------
    gemm_bf16_wmma<EPI_BF16, CC><<<dim3(NTOK / 256, QKVN / 64), 256, 0, stream>>>(
        xln, wqkvT, bqkv, nullptr, qkv, QKVN);

    // ---- window attention -------------------------------------------------
    attn_kernel<<<(NTOK / WTOK) * NH, 128, 0, stream>>>(qkv, attn);

    // ---- proj GEMM + window reverse + residual ----------------------------
    gemm_bf16_wmma<EPI_PROJ, CC><<<dim3(NTOK / 256, CC / 64), 256, 0, stream>>>(
        attn, wprojT, bproj, x, y, CC);

    // ---- LN2 --------------------------------------------------------------
    ln_kernel<false><<<NTOK / 8, 256, 0, stream>>>(y, g2, b2, ln2b);

    // ---- FC1 GEMM + GELU --------------------------------------------------
    gemm_bf16_wmma<EPI_GELU, CC><<<dim3(NTOK / 256, HID / 64), 256, 0, stream>>>(
        ln2b, w1T, bfc1, nullptr, hid, HID);

    // ---- FC2 GEMM + residual -> output ------------------------------------
    gemm_bf16_wmma<EPI_ADDF32, HID><<<dim3(NTOK / 256, CC / 64), 256, 0, stream>>>(
        hid, w2T, bfc2, y, out, CC);
}
